// CustomGINConv_31980326486101
// MI455X (gfx1250) — compile-verified, matrix operand order checked by reading, not run on hardware
//
#include <hip/hip_runtime.h>
#include <hip/hip_bf16.h>

typedef __attribute__((ext_vector_type(4)))  _Float16 v4h;
typedef __attribute__((ext_vector_type(8)))  _Float16 v8h;
typedef __attribute__((ext_vector_type(16))) _Float16 v16h;
typedef __attribute__((ext_vector_type(8)))  float    v8f;

#define CDIM 128
#define LDIM 6
#define TM   16          // rows of N handled per block
#define STRH 136         // LDS row stride in halfs (136/2 = 68 dwords == 4 mod 64 -> conflict-free b128)

// Stage 0: fold emb @ Wt[128:256,:] + bt into a 100x128 f32 table (tiny, L2-resident afterwards)
__global__ void embw_kernel(const float* __restrict__ emb, const float* __restrict__ Wt,
                            const float* __restrict__ bt, float* __restrict__ embW) {
    const int e = blockIdx.x;
    const int j = threadIdx.x;
    float s = bt[j];
    for (int k = 0; k < CDIM; ++k)
        s += emb[e * CDIM + k] * Wt[(CDIM + k) * CDIM + j];
    embW[e * CDIM + j] = s;
}

// Three independent accumulator chains (one per split-product) -> 3-way WMMA ILP
struct Acc3 { v8f a0, a1, a2; };

// Build v16h A operand for K-step i from two contiguous 8-half chunks (ISA 16-bit A 16x32 layout).
// base must already include the lane-half K offset (lh*8).
__device__ __forceinline__ v16h loadA(const _Float16* base, int i) {
    v8h a0 = *(const v8h*)(base + 32 * i);
    v8h a1 = *(const v8h*)(base + 32 * i + 16);
    return __builtin_shufflevector(a0, a1, 0,1,2,3,4,5,6,7,8,9,10,11,12,13,14,15);
}

// Load + split one 128x16 f32 weight slice into hi/lo f16 B operands (4 K-steps of 32).
__device__ __forceinline__ void loadB(const float* __restrict__ W, int n, int lh,
                                      v16h* Bh, v16h* Bl) {
    #pragma unroll
    for (int i = 0; i < 4; ++i) {
        #pragma unroll
        for (int e = 0; e < 16; ++e) {
            const int k = 32 * i + (e & 7) + ((e >> 3) * 16) + lh * 8; // K(e) per 16-bit operand layout
            const float wf = W[k * CDIM + n];
            const _Float16 wh = (_Float16)wf;
            Bh[i][e] = wh;
            Bl[i][e] = (_Float16)(wf - (float)wh);
        }
    }
}

// One 16x16 tile, K=128, split-precision: Ah*Bh + Al*Bh + Ah*Bl on 3 separate chains.
__device__ __forceinline__ void gemm_tile(const _Float16* Ah, const _Float16* Al,
                                          const v16h* Bh, const v16h* Bl, Acc3& acc) {
    #pragma unroll
    for (int i = 0; i < 4; ++i) {
        v16h ah = loadA(Ah, i);
        v16h al = loadA(Al, i);
        acc.a0 = __builtin_amdgcn_wmma_f32_16x16x32_f16(false, ah, false, Bh[i], (short)0, acc.a0, false, false);
        acc.a1 = __builtin_amdgcn_wmma_f32_16x16x32_f16(false, al, false, Bh[i], (short)0, acc.a1, false, false);
        acc.a2 = __builtin_amdgcn_wmma_f32_16x16x32_f16(false, ah, false, Bl[i], (short)0, acc.a2, false, false);
    }
}

__device__ __forceinline__ v8f acc_sum(const Acc3& acc) {
    return acc.a0 + acc.a1 + acc.a2;
}

__device__ __forceinline__ void split_store(_Float16* ph, _Float16* pl, int idx, float f) {
    const _Float16 h = (_Float16)f;
    ph[idx] = h;
    pl[idx] = (_Float16)(f - (float)h);
}

// Fully fused: h = x@WtA + embW[at]; shift-add; z=(1+eps)x+prop; g=relu(z@W1+b1); out=sum_l g@W2 + 6*b2
__global__ __launch_bounds__(256) void gin_fused(
    const float* __restrict__ x, const int* __restrict__ atype,
    const float* __restrict__ Wt, const float* __restrict__ eps,
    const float* __restrict__ W1, const float* __restrict__ b1,
    const float* __restrict__ W2, const float* __restrict__ b2,
    const float* __restrict__ embW, float* __restrict__ out, int N)
{
    __shared__ _Float16 xs_h[LDIM * TM * STRH];
    __shared__ _Float16 xs_l[LDIM * TM * STRH];
    __shared__ _Float16 hs_h[LDIM * TM * STRH];
    __shared__ _Float16 hs_l[LDIM * TM * STRH];
    __shared__ int      atl[LDIM * TM];

    const int tid  = threadIdx.x;
    const int wave = tid >> 5;            // 0..7 -> output column tile
    const int lane = tid & 31;
    const int m    = lane & 15;           // A row / D column-within-tile
    const int lh   = lane >> 4;           // lane half
    const int n    = wave * 16 + m;       // output column this lane produces
    const int rowBase = blockIdx.x * TM;

    // ---- stage x tile into LDS, split f32 -> f16 hi/lo (x read exactly once from HBM) ----
    for (int i = tid; i < LDIM * TM * (CDIM / 4); i += 256) {
        const int l  = i / (TM * (CDIM / 4));
        const int r  = (i / (CDIM / 4)) % TM;
        const int c4 = i % (CDIM / 4);
        const float4 v = ((const float4*)x)[(size_t)(l * N + rowBase + r) * (CDIM / 4) + c4];
        v4h hh, ll;
        hh[0] = (_Float16)v.x; ll[0] = (_Float16)(v.x - (float)hh[0]);
        hh[1] = (_Float16)v.y; ll[1] = (_Float16)(v.y - (float)hh[1]);
        hh[2] = (_Float16)v.z; ll[2] = (_Float16)(v.z - (float)hh[2]);
        hh[3] = (_Float16)v.w; ll[3] = (_Float16)(v.w - (float)hh[3]);
        const int off = (l * TM + r) * STRH + c4 * 4;
        *(v4h*)&xs_h[off] = hh;
        *(v4h*)&xs_l[off] = ll;
    }
    if (tid < LDIM * TM) {
        const int l = tid / TM, r = tid % TM;
        atl[tid] = atype[l * N + rowBase + r];
    }
    __syncthreads();

    const int aBase = m * STRH + lh * 8;  // per-lane A row base (add l*TM*STRH per layer)

    // ---- GEMM1: h[l] = x[l] @ Wt[:128,:] + embW[at]  (embW already includes bt) ----
    {
        v16h Bh[4], Bl[4];
        loadB(Wt, n, lh, Bh, Bl);
        for (int l = 0; l < LDIM; ++l) {
            Acc3 acc = {};
            gemm_tile(&xs_h[l * TM * STRH + aBase], &xs_l[l * TM * STRH + aBase], Bh, Bl, acc);
            const v8f r = acc_sum(acc);
            #pragma unroll
            for (int v = 0; v < 8; ++v) {
                const int row = v + lh * 8;
                const float f = r[v] + embW[atl[l * TM + row] * CDIM + n];
                split_store(hs_h, hs_l, (l * TM + row) * STRH + n, f);
            }
        }
    }
    __syncthreads();

    // ---- shift-add along path dim + GIN residual, in place over xs planes ----
    {
        const float epsv = 1.0f + eps[0];
        for (int i = tid; i < LDIM * TM * CDIM; i += 256) {
            const int l = i / (TM * CDIM);
            const int r = (i / CDIM) % TM;
            const int c = i % CDIM;
            const int idx = (l * TM + r) * STRH + c;
            float p = 0.0f;
            if (l > 0)        p += (float)hs_h[idx - TM * STRH] + (float)hs_l[idx - TM * STRH];
            if (l < LDIM - 1) p += (float)hs_h[idx + TM * STRH] + (float)hs_l[idx + TM * STRH];
            const float xv = (float)xs_h[idx] + (float)xs_l[idx];
            split_store(xs_h, xs_l, idx, epsv * xv + p);
        }
    }
    __syncthreads();

    // ---- GEMM2: g[l] = relu(z[l] @ W1 + b1), stored into hs planes ----
    {
        v16h Bh[4], Bl[4];
        loadB(W1, n, lh, Bh, Bl);
        const float b1v = b1[n];
        for (int l = 0; l < LDIM; ++l) {
            Acc3 acc = {};
            gemm_tile(&xs_h[l * TM * STRH + aBase], &xs_l[l * TM * STRH + aBase], Bh, Bl, acc);
            const v8f r = acc_sum(acc);
            #pragma unroll
            for (int v = 0; v < 8; ++v) {
                const int row = v + lh * 8;
                const float g = fmaxf(r[v] + b1v, 0.0f);
                split_store(hs_h, hs_l, (l * TM + row) * STRH + n, g);
            }
        }
    }
    __syncthreads();

    // ---- GEMM3: out = sum_l g[l] @ W2 + 6*b2, three chains carried across all layers ----
    {
        v16h Bh[4], Bl[4];
        loadB(W2, n, lh, Bh, Bl);
        Acc3 acc = {};
        for (int l = 0; l < LDIM; ++l)
            gemm_tile(&hs_h[l * TM * STRH + aBase], &hs_l[l * TM * STRH + aBase], Bh, Bl, acc);
        const v8f r = acc_sum(acc);
        const float b2v = b2[n] * (float)LDIM;
        #pragma unroll
        for (int v = 0; v < 8; ++v) {
            const int row = v + lh * 8;
            out[(size_t)(rowBase + row) * CDIM + n] = r[v] + b2v;
        }
    }
}

extern "C" void kernel_launch(void* const* d_in, const int* in_sizes, int n_in,
                              void* d_out, int out_size, void* d_ws, size_t ws_size,
                              hipStream_t stream) {
    const float* x    = (const float*)d_in[0];
    const int*   at   = (const int*)d_in[1];
    const float* emb  = (const float*)d_in[2];
    const float* Wt   = (const float*)d_in[3];
    const float* bt   = (const float*)d_in[4];
    const float* eps  = (const float*)d_in[5];
    const float* W1   = (const float*)d_in[6];
    const float* b1   = (const float*)d_in[7];
    const float* W2   = (const float*)d_in[8];
    const float* b2   = (const float*)d_in[9];
    float* out  = (float*)d_out;
    float* embW = (float*)d_ws;                 // 100*128 floats of scratch

    const int N = in_sizes[0] / (LDIM * CDIM);  // 65536

    embw_kernel<<<dim3(100), dim3(CDIM), 0, stream>>>(emb, Wt, bt, embW);
    gin_fused<<<dim3(N / TM), dim3(256), 0, stream>>>(x, at, Wt, eps, W1, b1, W2, b2,
                                                      embW, out, N);
}
